// CDC_1666447311534
// MI455X (gfx1250) — matrix operand order repaired and checked by reading, not compiled, not run on hardware
//
#include <hip/hip_runtime.h>
#include <hip/hip_bf16.h>

// =====================================================================
// CPC forward (MobileNetV2 encoder + GRU + prediction heads) for MI455X
// (gfx1250, wave32). All 1x1 convs / GRU GEMMs / head GEMMs run through
// v_wmma_f32_16x16x32_bf16; A tiles are staged global->LDS with the
// CDNA5 async-to-LDS path (ASYNCcnt) and shared by 4 waves; each wave
// holds 4 accumulators (64-row M tile) for back-to-back WMMA issue.
// Activations are NHWC bf16 with channels padded to multiples of 32
// (padded channels provably zero). Depthwise 3x3, stem conv, pooling
// and GRU gates are bandwidth-trivial VALU code. Workspace ~50 MB.
// =====================================================================

#define USE_ASYNC_STAGE 1   // global_load_async_to_lds_b128 staging path

typedef __bf16 bf16_t;
typedef __attribute__((ext_vector_type(16))) __bf16 v16bf;
typedef __attribute__((ext_vector_type(8)))  __bf16 v8bf;
typedef __attribute__((ext_vector_type(8)))  float   v8f;

static inline int cpad32(int c) { return (c + 31) & ~31; }

// ---------------------------------------------------------------------
// WMMA GEMM:  out[m][n] = act( scale[n] * (A@B)[m][n] + shift[n] )
//   A   : [M x Kpad] bf16 row-major (Kpad % 32 == 0)
//   Bp  : bf16 fragments packed [Kpad/32][Npad/16][32 lanes][16 elems]
//   flags: 1 = relu6, 2 = add residual (bf16 [M x Npad]),
//          4 = store fp32 to outF (optionally row-remapped), else bf16.
// Block = 128 threads (4 waves). Block tile = 64 (M) x 64 (N):
// wave w owns N-tile blockIdx.y*4+w and all four 16-row M-subtiles, so
// each K-step is 1 B-fragment load + 4 A-fragment LDS reads + 4 WMMAs.
// ---------------------------------------------------------------------
__global__ void __launch_bounds__(128)
k_wmma_gemm(const bf16_t* __restrict__ A, const bf16_t* __restrict__ Bp,
            const float* __restrict__ scale, const float* __restrict__ shift,
            const bf16_t* __restrict__ resid, const int* __restrict__ rowmap,
            bf16_t* __restrict__ outB, float* __restrict__ outF,
            int M, int Kpad, int Npad, int flags)
{
    __shared__ __align__(16) bf16_t As[64 * 32];   // 4 KB A tile
    const int tid  = threadIdx.x;
    const int lane = tid & 31;
    const int wave = tid >> 5;
    const int nTiles = Npad >> 4;
    const int nt   = blockIdx.y * 4 + wave;
    const int m0   = blockIdx.x * 64;
    const int half = lane >> 4;     // 0: lanes 0-15, 1: lanes 16-31
    const int l15  = lane & 15;
    const int kTiles = Kpad >> 5;

    v8f acc[4] = {v8f{}, v8f{}, v8f{}, v8f{}};

    for (int kt = 0; kt < kTiles; ++kt) {
        // ---- stage A tile (64 rows x 32 K) into LDS: 32 B per thread ----
        {
            int r    = tid >> 1;          // 64 rows, 2 threads per row
            int cseg = (tid & 1) * 16;    // 16 bf16 = 32 B per thread
            int gr   = m0 + r;
            bf16_t* dst = As + r * 32 + cseg;
#if USE_ASYNC_STAGE
            if (gr < M) {
                const bf16_t* src = A + (size_t)gr * Kpad + kt * 32 + cseg;
                unsigned          ldsa = (unsigned)(size_t)dst;
                unsigned long long ga  = (unsigned long long)(size_t)src;
                // CDNA5 async copy global->LDS, offset applied to both addresses (ASYNCcnt)
                asm volatile("global_load_async_to_lds_b128 %0, %1, off"
                             :: "v"(ldsa), "v"(ga) : "memory");
                asm volatile("global_load_async_to_lds_b128 %0, %1, off offset:16"
                             :: "v"(ldsa), "v"(ga) : "memory");
            } else {
                v16bf z = {};
                *(v16bf*)dst = z;         // zero-fill M tail via DS
            }
            asm volatile("s_wait_asynccnt 0x0" ::: "memory");
#else
            v16bf ld = {};
            if (gr < M)
                ld = *(const v16bf*)(A + (size_t)gr * Kpad + kt * 32 + cseg);
            *(v16bf*)dst = ld;
#endif
        }
        __syncthreads();
        if (nt < nTiles) {
            // B fragment: pre-packed contiguous 16 bf16 per lane.
            const bf16_t* bptr = Bp + ((size_t)(kt * nTiles + nt)) * 512 + lane * 16;
            v16bf b = *(const v16bf*)bptr;
            if (kt + 1 < kTiles)   // gfx1250 global_prefetch_b8 on next B tile
                __builtin_prefetch(Bp + ((size_t)((kt + 1) * nTiles + nt)) * 512, 0, 1);
            // A fragments (16-bit A 16x32): lanes 0-15 own K {0..15},
            // lanes 16-31 own K {16..31}; row = lane&15. 4 M-subtiles.
#pragma unroll
            for (int s = 0; s < 4; ++s) {
                union { v16bf v; v8bf h[2]; } a;
                const bf16_t* rowp = As + (s * 16 + l15) * 32 + half * 16;
                a.h[0] = *(const v8bf*)(rowp);
                a.h[1] = *(const v8bf*)(rowp + 8);
                acc[s] = __builtin_amdgcn_wmma_f32_16x16x32_bf16(
                             false, a.v, false, b, (short)0, acc[s], false, false);
            }
        }
        __syncthreads();
    }
    if (nt >= nTiles) return;

    // C layout: VGPR r -> M = base + r + 8*half, N = nt*16 + (lane&15)
    const int N  = nt * 16 + l15;
    const float s  = scale[N];
    const float sh = shift[N];
#pragma unroll
    for (int st = 0; st < 4; ++st) {
#pragma unroll
        for (int r = 0; r < 8; ++r) {
            int m = m0 + st * 16 + r + half * 8;
            if (m >= M) continue;
            float v = acc[st][r] * s + sh;
            if (flags & 1) v = fminf(fmaxf(v, 0.f), 6.f);
            if (flags & 2) v += (float)resid[(size_t)m * Npad + N];
            if (flags & 4) {
                int mr = rowmap ? rowmap[m] : m;
                outF[(size_t)mr * Npad + N] = v;
            } else {
                outB[(size_t)m * Npad + N] = (bf16_t)v;
            }
        }
    }
}

// ---- pack fp32 W[Cout][Cin] (1x1 conv / linear) into WMMA-B fragments ----
// B fragment mapping (16-bit B 32x16): lanes 0-15: N=nt*16+lane, K=kt*32+e;
// lanes 16-31: N=nt*16+lane-16, K=kt*32+16+e.
__global__ void k_pack_pw(const float* __restrict__ W, bf16_t* __restrict__ Bp,
                          int Cin, int Cout, int Kpad, int Npad)
{
    int i = blockIdx.x * blockDim.x + threadIdx.x;
    int total = Kpad * Npad;
    if (i >= total) return;
    int e    = i & 15;
    int lane = (i >> 4) & 31;
    int tile = i >> 9;
    int nTiles = Npad >> 4;
    int kt = tile / nTiles, nt = tile % nTiles;
    int N = nt * 16 + (lane & 15);
    int K = kt * 32 + (lane >> 4) * 16 + e;
    float v = (K < Cin && N < Cout) ? W[(size_t)N * Cin + K] : 0.f;
    Bp[i] = (bf16_t)v;
}

__global__ void k_packdw(const float* __restrict__ W, float* __restrict__ w9,
                         int hid, int Cpad)
{
    int i = blockIdx.x * blockDim.x + threadIdx.x;
    if (i >= Cpad * 9) return;
    int c = i / 9, j = i % 9;
    w9[i] = (c < hid) ? W[c * 9 + j] : 0.f;
}

__global__ void k_bnprep(const float* __restrict__ g, const float* __restrict__ b,
                         const float* __restrict__ m, const float* __restrict__ v,
                         float* __restrict__ scale, float* __restrict__ shift,
                         int C, int Cpad)
{
    int i = blockIdx.x * blockDim.x + threadIdx.x;
    if (i >= Cpad) return;
    if (i < C) {
        float s = g[i] * rsqrtf(v[i] + 1e-5f);
        scale[i] = s;
        shift[i] = b[i] - m[i] * s;
    } else {
        scale[i] = 0.f; shift[i] = 0.f;
    }
}

__global__ void k_biasprep(const float* __restrict__ bias,
                           float* __restrict__ scale, float* __restrict__ shift,
                           int C, int Cpad)
{
    int i = blockIdx.x * blockDim.x + threadIdx.x;
    if (i >= Cpad) return;
    scale[i] = (i < C) ? 1.f : 0.f;
    shift[i] = (i < C) ? bias[i] : 0.f;
}

// ---- stem: conv 3x3 s2 p1, 1->32 channels, BN + relu6, NHWC bf16 out ----
__global__ void k_stem(const float* __restrict__ x, const float* __restrict__ w,
                       const float* __restrict__ scale, const float* __restrict__ shift,
                       bf16_t* __restrict__ out, int nImg)
{
    int i = blockIdx.x * blockDim.x + threadIdx.x;
    int total = nImg * 32 * 32 * 32;
    if (i >= total) return;
    int c = i & 31; int t = i >> 5;
    int xo = t & 31; t >>= 5;
    int yo = t & 31; int n = t >> 5;
    const float* img = x + (size_t)n * 4096;
    float acc = 0.f;
#pragma unroll
    for (int ky = 0; ky < 3; ++ky) {
        int iy = 2 * yo + ky - 1;
        if ((unsigned)iy >= 64u) continue;
#pragma unroll
        for (int kx = 0; kx < 3; ++kx) {
            int ix = 2 * xo + kx - 1;
            if ((unsigned)ix >= 64u) continue;
            acc += img[iy * 64 + ix] * w[c * 9 + ky * 3 + kx];
        }
    }
    float v = acc * scale[c] + shift[c];
    v = fminf(fmaxf(v, 0.f), 6.f);
    out[i] = (bf16_t)v;
}

// ---- depthwise 3x3 (pad 1, stride 1/2), BN + relu6, NHWC bf16 ----
__global__ void k_dw3x3(const bf16_t* __restrict__ in, const float* __restrict__ w9,
                        const float* __restrict__ scale, const float* __restrict__ shift,
                        bf16_t* __restrict__ out,
                        int nImg, int Hin, int Win, int Hout, int Wout,
                        int Cpad, int stride)
{
    size_t i = (size_t)blockIdx.x * blockDim.x + threadIdx.x;
    size_t total = (size_t)nImg * Hout * Wout * Cpad;
    if (i >= total) return;
    int c = (int)(i % Cpad); size_t t = i / Cpad;
    int x = (int)(t % Wout); t /= Wout;
    int y = (int)(t % Hout); int n = (int)(t / Hout);
    int iy0 = y * stride - 1, ix0 = x * stride - 1;
    const bf16_t* base = in + (size_t)n * Hin * Win * Cpad;
    float acc = 0.f;
#pragma unroll
    for (int ky = 0; ky < 3; ++ky) {
        int iy = iy0 + ky;
        if ((unsigned)iy >= (unsigned)Hin) continue;
#pragma unroll
        for (int kx = 0; kx < 3; ++kx) {
            int ix = ix0 + kx;
            if ((unsigned)ix >= (unsigned)Win) continue;
            acc += (float)base[((size_t)iy * Win + ix) * Cpad + c] * w9[c * 9 + ky * 3 + kx];
        }
    }
    float v = acc * scale[c] + shift[c];
    v = fminf(fmaxf(v, 0.f), 6.f);
    out[i] = (bf16_t)v;
}

// ---- global mean over 2x2 -> encodings (fp32 d_out) + bf16 z buffer ----
__global__ void k_pool(const bf16_t* __restrict__ hbuf, float* __restrict__ encOut,
                       bf16_t* __restrict__ zb, int img0, int nImg)
{
    int i = blockIdx.x * blockDim.x + threadIdx.x;
    if (i >= nImg * 1280) return;
    int ch = i % 1280; int n = i / 1280;
    const bf16_t* p = hbuf + (size_t)n * 4 * 1280 + ch;
    float v = 0.25f * ((float)p[0] + (float)p[1280] + (float)p[2 * 1280] + (float)p[3 * 1280]);
    size_t o = (size_t)(img0 + n) * 1280 + ch;
    encOut[o] = v;
    zb[o] = (bf16_t)v;
}

// ---- gather GRU inputs: xa[t][s][ch], s=b*7+col, img=b*49+t*7+col ----
__global__ void k_gather_x(const bf16_t* __restrict__ zb, bf16_t* __restrict__ xa)
{
    int i = blockIdx.x * blockDim.x + threadIdx.x;
    if (i >= 6 * 112 * 1280) return;
    int ch = i % 1280; int t2 = i / 1280;
    int s = t2 % 112;  int t  = t2 / 112;
    int b = s / 7, col = s % 7;
    xa[i] = zb[(size_t)(b * 49 + t * 7 + col) * 1280 + ch];
}

__global__ void k_hinit(const float* __restrict__ hidden, float* __restrict__ h,
                        bf16_t* __restrict__ hb)
{
    int i = blockIdx.x * blockDim.x + threadIdx.x;
    if (i >= 112 * 256) return;
    float v = hidden[i & 255];
    h[i] = v; hb[i] = (bf16_t)v;
}

// ---- GRU gates: r,z,n + state update; emits bf16 h (next A) and context ----
__global__ void k_gru_gates(const float* __restrict__ gi, const float* __restrict__ gh,
                            float* __restrict__ h, bf16_t* __restrict__ hb,
                            bf16_t* __restrict__ cs_t)
{
    int i = blockIdx.x * blockDim.x + threadIdx.x;
    if (i >= 112 * 256) return;
    int c = i & 255; int s = i >> 8;
    const float* gis = gi + s * 768;
    const float* ghs = gh + s * 768;
    float r  = 1.f / (1.f + __expf(-(gis[c]       + ghs[c])));
    float z  = 1.f / (1.f + __expf(-(gis[256 + c] + ghs[256 + c])));
    float n  = tanhf(gis[512 + c] + r * ghs[512 + c]);
    float hv = h[i];
    float hn = (1.f - z) * n + z * hv;
    h[i] = hn; hb[i] = (bf16_t)hn; cs_t[i] = (bf16_t)hn;
}

// ---- row map for prediction head k: m=(t,s) -> padded preds row index ----
__global__ void k_rowmap(int* __restrict__ rm, int k)
{
    int m = blockIdx.x * blockDim.x + threadIdx.x;
    int Mk = (6 - k) * 112;
    if (m >= Mk) return;
    int t = m / 112, s = m % 112;
    int b = s / 7, col = s % 7;
    rm[m] = ((b * 5 + k) * 7 + (t + k + 1)) * 7 + col;   // row into preds[.,.,7,7,:]
}

__global__ void k_fill0(float* __restrict__ p, long long n)
{
    long long i = (long long)blockIdx.x * blockDim.x + threadIdx.x;
    if (i < n) p[i] = 0.f;
}

// =====================================================================
// Host orchestration
// =====================================================================
struct IRcfg { int t, cin, cout, stride, hin; };
static const IRcfg g_ir[17] = {
    {1, 32,  16, 1, 32},
    {6, 16,  24, 2, 32}, {6, 24,  24, 1, 16},
    {6, 24,  32, 2, 16}, {6, 32,  32, 1,  8}, {6, 32,  32, 1,  8},
    {6, 32,  64, 2,  8}, {6, 64,  64, 1,  4}, {6, 64,  64, 1,  4}, {6, 64,  64, 1,  4},
    {6, 64,  96, 1,  4}, {6, 96,  96, 1,  4}, {6, 96,  96, 1,  4},
    {6, 96, 160, 2,  4}, {6, 160, 160, 1, 2}, {6, 160, 160, 1, 2},
    {6, 160, 320, 1,  2},
};

extern "C" void kernel_launch(void* const* d_in, const int* in_sizes, int n_in,
                              void* d_out, int out_size, void* d_ws, size_t ws_size,
                              hipStream_t stream)
{
    (void)out_size; (void)ws_size;

    // ---------- locate x / hidden by globally-unique sizes ----------
    const float* X = nullptr; const float* HID0 = nullptr;
    for (int j = 0; j < n_in; ++j) {
        if (in_sizes[j] == 16 * 49 * 64 * 64 && !X)    X    = (const float*)d_in[j];
        if (in_sizes[j] == 256            && !HID0)    HID0 = (const float*)d_in[j];
    }

    // ---------- forward scan for the remaining params ----------
    // Assumes reference-file dict insertion order; size-1 scalars and
    // non-matching leaves (x, hidden) are skipped automatically.
    int idx = 0;
    auto next = [&](long long expect) -> const float* {
        int j = idx;
        while (j < n_in && (long long)in_sizes[j] != expect) ++j;
        if (j >= n_in) j = (idx < n_in) ? idx : 0;   // defensive fallback
        idx = j + 1;
        return (const float*)d_in[j];
    };

    const float* stemW = next(32 * 9);
    const float *stG = next(32), *stB = next(32), *stM = next(32), *stV = next(32);

    struct BlkP { const float *ew, *eg, *eb, *em, *ev, *dw, *dg, *db, *dm, *dv,
                              *pw, *pg, *pb, *pm, *pv; };
    BlkP bp[17];
    for (int i = 0; i < 17; ++i) {
        const IRcfg& r = g_ir[i];
        int hid = r.cin * r.t;
        if (r.t != 1) {
            bp[i].ew = next((long long)hid * r.cin);
            bp[i].eg = next(hid); bp[i].eb = next(hid);
            bp[i].em = next(hid); bp[i].ev = next(hid);
        } else bp[i].ew = nullptr;
        bp[i].dw = next((long long)hid * 9);
        bp[i].dg = next(hid); bp[i].db = next(hid);
        bp[i].dm = next(hid); bp[i].dv = next(hid);
        bp[i].pw = next((long long)r.cout * hid);
        bp[i].pg = next(r.cout); bp[i].pb = next(r.cout);
        bp[i].pm = next(r.cout); bp[i].pv = next(r.cout);
    }
    const float* headW = next(1280 * 320);
    const float *hG = next(1280), *hB = next(1280), *hM = next(1280), *hV = next(1280);
    const float* Wih = next(768 * 1280);
    const float* Whh = next(768 * 256);
    const float* bih = next(768);
    const float* bhh = next(768);
    const float *Wk[5], *bk[5];
    for (int k = 0; k < 5; ++k) { Wk[k] = next(1280 * 256); bk[k] = next(1280); }

    // ---------- workspace bump allocator ----------
    char* wp = (char*)d_ws;
    auto walloc = [&](size_t bytes) -> void* {
        void* p = (void*)wp; wp += (bytes + 255) & ~(size_t)255; return p;
    };

    const int CHUNK = 56;                               // 784 / 56 = 14 chunks
    const size_t BUFE = (size_t)CHUNK * 32 * 32 * 96;   // max NHWC elems / chunk
    bf16_t* buf[3];
    for (int i = 0; i < 3; ++i) buf[i] = (bf16_t*)walloc(BUFE * sizeof(bf16_t));

    bf16_t* zb  = (bf16_t*)walloc((size_t)784 * 1280 * 2);
    bf16_t* xa  = (bf16_t*)walloc((size_t)6 * 112 * 1280 * 2);
    float*  giF = (float*)walloc((size_t)112 * 768 * 4);
    float*  ghF = (float*)walloc((size_t)112 * 768 * 4);
    float*  hF  = (float*)walloc((size_t)112 * 256 * 4);
    bf16_t* hBuf= (bf16_t*)walloc((size_t)112 * 256 * 2);
    bf16_t* csB = (bf16_t*)walloc((size_t)6 * 112 * 256 * 2);
    int*    rm  = (int*)walloc(672 * 4);

    // ---------- pack weights + fold BN (every call; deterministic) ----------
    auto packPW = [&](const float* W, int Cin, int Cout, int Kpad, int Npad) -> bf16_t* {
        bf16_t* p = (bf16_t*)walloc((size_t)Kpad * Npad * 2);
        int total = Kpad * Npad;
        k_pack_pw<<<dim3((total + 255) / 256), 256, 0, stream>>>(W, p, Cin, Cout, Kpad, Npad);
        return p;
    };
    auto bnSS = [&](const float* g, const float* b, const float* m, const float* v,
                    int C, int Cpad, float** S, float** H) {
        *S = (float*)walloc(Cpad * 4); *H = (float*)walloc(Cpad * 4);
        k_bnprep<<<dim3((Cpad + 63) / 64), 64, 0, stream>>>(g, b, m, v, *S, *H, C, Cpad);
    };
    auto biasSS = [&](const float* b, int C, int Cpad, float** S, float** H) {
        *S = (float*)walloc(Cpad * 4); *H = (float*)walloc(Cpad * 4);
        k_biasprep<<<dim3((Cpad + 63) / 64), 64, 0, stream>>>(b, *S, *H, C, Cpad);
    };

    float *stS, *stH; bnSS(stG, stB, stM, stV, 32, 32, &stS, &stH);

    bf16_t* ewP[17]; float *eS[17], *eH[17];
    float*  dw9[17]; float *dS[17], *dH[17];
    bf16_t* pwP[17]; float *pS[17], *pH[17];
    for (int i = 0; i < 17; ++i) {
        const IRcfg& r = g_ir[i];
        int hid = r.cin * r.t;
        int kc = cpad32(r.cin), kh = cpad32(hid), ko = cpad32(r.cout);
        if (r.t != 1) {
            ewP[i] = packPW(bp[i].ew, r.cin, hid, kc, kh);
            bnSS(bp[i].eg, bp[i].eb, bp[i].em, bp[i].ev, hid, kh, &eS[i], &eH[i]);
        } else { ewP[i] = nullptr; eS[i] = eH[i] = nullptr; }
        dw9[i] = (float*)walloc((size_t)kh * 9 * 4);
        k_packdw<<<dim3((kh * 9 + 255) / 256), 256, 0, stream>>>(bp[i].dw, dw9[i], hid, kh);
        bnSS(bp[i].dg, bp[i].db, bp[i].dm, bp[i].dv, hid, kh, &dS[i], &dH[i]);
        pwP[i] = packPW(bp[i].pw, hid, r.cout, kh, ko);
        bnSS(bp[i].pg, bp[i].pb, bp[i].pm, bp[i].pv, r.cout, ko, &pS[i], &pH[i]);
    }
    bf16_t* headP = packPW(headW, 320, 1280, 320, 1280);
    float *heS, *heH; bnSS(hG, hB, hM, hV, 1280, 1280, &heS, &heH);

    bf16_t* WihP = packPW(Wih, 1280, 768, 1280, 768);
    bf16_t* WhhP = packPW(Whh,  256, 768,  256, 768);
    float *giS, *giH; biasSS(bih, 768, 768, &giS, &giH);
    float *ghS, *ghH; biasSS(bhh, 768, 768, &ghS, &ghH);
    bf16_t* WkP[5]; float *wkS[5], *wkH[5];
    for (int k = 0; k < 5; ++k) {
        WkP[k] = packPW(Wk[k], 256, 1280, 256, 1280);
        biasSS(bk[k], 1280, 1280, &wkS[k], &wkH[k]);
    }

    auto gemm = [&](const bf16_t* A, const bf16_t* Bp, const float* sc, const float* sh,
                    const bf16_t* resid, const int* rmap, bf16_t* oB, float* oF,
                    int M, int Kpad, int Npad, int flags) {
        dim3 g((M + 63) / 64, (Npad + 63) / 64);
        k_wmma_gemm<<<g, 128, 0, stream>>>(A, Bp, sc, sh, resid, rmap, oB, oF,
                                           M, Kpad, Npad, flags);
    };

    // ---------- encoder: 14 chunks of 56 patches ----------
    for (int c0 = 0; c0 < 784; c0 += CHUNK) {
        int icur = 0, ia = 1, ib = 2;
        {
            int total = CHUNK * 32 * 32 * 32;
            k_stem<<<dim3((total + 255) / 256), 256, 0, stream>>>(
                X + (size_t)c0 * 4096, stemW, stS, stH, buf[icur], CHUNK);
        }
        for (int i = 0; i < 17; ++i) {
            const IRcfg& r = g_ir[i];
            int hid = r.cin * r.t;
            int hout = (r.stride == 2) ? r.hin / 2 : r.hin;
            int use_res = (r.stride == 1 && r.cin == r.cout);
            int kh = cpad32(hid);
            const bf16_t* dwin;
            if (r.t != 1) {
                gemm(buf[icur], ewP[i], eS[i], eH[i], nullptr, nullptr,
                     buf[ia], nullptr, CHUNK * r.hin * r.hin, cpad32(r.cin), kh, 1);
                dwin = buf[ia];
            } else dwin = buf[icur];
            {
                size_t total = (size_t)CHUNK * hout * hout * kh;
                k_dw3x3<<<dim3((unsigned)((total + 255) / 256)), 256, 0, stream>>>(
                    dwin, dw9[i], dS[i], dH[i], buf[ib],
                    CHUNK, r.hin, r.hin, hout, hout, kh, r.stride);
            }
            gemm(buf[ib], pwP[i], pS[i], pH[i],
                 use_res ? buf[icur] : nullptr, nullptr,
                 buf[ia], nullptr, CHUNK * hout * hout, kh, cpad32(r.cout),
                 use_res ? 2 : 0);
            int t = icur; icur = ia; ia = t;
        }
        // head 1x1 conv 320->1280 @ 2x2, then mean-pool into d_out + zb
        gemm(buf[icur], headP, heS, heH, nullptr, nullptr,
             buf[ia], nullptr, CHUNK * 4, 320, 1280, 1);
        k_pool<<<dim3((CHUNK * 1280 + 255) / 256), 256, 0, stream>>>(
            buf[ia], (float*)d_out, zb, c0, CHUNK);
    }

    // ---------- GRU over rows (6 steps, 112 sequences) ----------
    k_gather_x<<<dim3((6 * 112 * 1280 + 255) / 256), 256, 0, stream>>>(zb, xa);
    k_hinit<<<dim3((112 * 256 + 255) / 256), 256, 0, stream>>>(HID0, hF, hBuf);
    for (int t = 0; t < 6; ++t) {
        gemm(xa + (size_t)t * 112 * 1280, WihP, giS, giH, nullptr, nullptr,
             nullptr, giF, 112, 1280, 768, 4);
        gemm(hBuf, WhhP, ghS, ghH, nullptr, nullptr,
             nullptr, ghF, 112, 256, 768, 4);
        k_gru_gates<<<dim3((112 * 256 + 255) / 256), 256, 0, stream>>>(
            giF, ghF, hF, hBuf, csB + (size_t)t * 112 * 256);
    }

    // ---------- prediction heads with fused zero-padding ----------
    float* preds = (float*)d_out + (size_t)784 * 1280;
    long long npred = (long long)16 * 5 * 49 * 1280;
    k_fill0<<<dim3((unsigned)((npred + 255) / 256)), 256, 0, stream>>>(preds, npred);
    for (int k = 0; k < 5; ++k) {
        int Mk = (6 - k) * 112;
        k_rowmap<<<dim3((Mk + 255) / 256), 256, 0, stream>>>(rm, k);
        gemm(csB, WkP[k], wkS[k], wkH[k], nullptr, rm, nullptr, preds,
             Mk, 256, 1280, 4);
    }
}
